// MOEGate_1657857376777
// MI455X (gfx1250) — matrix operand order repaired and checked by reading, not compiled, not run on hardware
//
#include <hip/hip_runtime.h>
#include <hip/hip_bf16.h>

#define TOKENS   16384
#define DIM      2048
#define NEXP     64
#define TOPK     8
#define SEQ      4096
#define NBATCH   4
#define TOK_PER_BLOCK 128   // 8 waves x 16 tokens

typedef __attribute__((ext_vector_type(16))) __bf16 v16bf;
typedef __attribute__((ext_vector_type(8)))  float  v8f;

union BF16x16 { unsigned int u[8]; v16bf v; };

__device__ __forceinline__ unsigned int pack_bf16(float lo, float hi) {
    unsigned int ul = __builtin_bit_cast(unsigned int, lo);
    unsigned int uh = __builtin_bit_cast(unsigned int, hi);
    return (ul >> 16) | (uh & 0xFFFF0000u);
}

// ---------------------------------------------------------------- prep ------
// Convert W [64][2048] f32 -> bf16 (row-major) into ws; zero 4*64 accumulators.
__global__ void moegate_prep(const float* __restrict__ W,
                             unsigned short* __restrict__ wbf,
                             float* __restrict__ ws_acc) {
    int i = blockIdx.x * blockDim.x + threadIdx.x;
    if (i < NBATCH * NEXP) ws_acc[i] = 0.0f;
    if (i < NEXP * DIM)
        wbf[i] = (unsigned short)(__builtin_bit_cast(unsigned int, W[i]) >> 16);
}

// ---------------------------------------------------------------- gate ------
__global__ void __launch_bounds__(256)
moegate_main(const float* __restrict__ h,
             const unsigned short* __restrict__ wbf,
             int* __restrict__ ids_out,
             float* __restrict__ w_out,
             float* __restrict__ ws_acc) {
    __shared__ float lds_logits[TOK_PER_BLOCK][NEXP + 1]; // +1 pad: bank spread
    __shared__ float blocksum[NEXP];

    const int tid  = threadIdx.x;
    const int lane = tid & 31;
    const int wave = tid >> 5;

    if (tid < NEXP) blocksum[tid] = 0.0f;

    // ---- Phase 1: logits via WMMA (uniform control flow, EXEC all ones) ----
    const int tokenBase = blockIdx.x * TOK_PER_BLOCK + wave * 16;
    const int sub   = lane & 15;
    const int hiHalf = lane >> 4;            // 0: lanes 0-15, 1: lanes 16-31
    const int kA = hiHalf * 16;              // A: K offsets 0..15 / 16..31
    const int kB = hiHalf * 16;              // B: K offsets 0..15 / 16..31

    const float* hrow = h + (size_t)(tokenBase + sub) * DIM;
    const unsigned short* w0 = wbf + (size_t)(0 * 16 + sub) * DIM;
    const unsigned short* w1 = wbf + (size_t)(1 * 16 + sub) * DIM;
    const unsigned short* w2 = wbf + (size_t)(2 * 16 + sub) * DIM;
    const unsigned short* w3 = wbf + (size_t)(3 * 16 + sub) * DIM;

    v8f acc0 = {}, acc1 = {}, acc2 = {}, acc3 = {};

    #pragma unroll 2
    for (int k0 = 0; k0 < DIM; k0 += 32) {
        // A fragment: 16 bf16 from one contiguous 16-float run
        float4 a0 = *(const float4*)(hrow + k0 + kA);
        float4 a1 = *(const float4*)(hrow + k0 + kA + 4);
        float4 a2 = *(const float4*)(hrow + k0 + kA + 8);
        float4 a3 = *(const float4*)(hrow + k0 + kA + 12);
        BF16x16 af;
        af.u[0] = pack_bf16(a0.x, a0.y); af.u[1] = pack_bf16(a0.z, a0.w);
        af.u[2] = pack_bf16(a1.x, a1.y); af.u[3] = pack_bf16(a1.z, a1.w);
        af.u[4] = pack_bf16(a2.x, a2.y); af.u[5] = pack_bf16(a2.z, a2.w);
        af.u[6] = pack_bf16(a3.x, a3.y); af.u[7] = pack_bf16(a3.z, a3.w);

        // B fragments: 16 contiguous bf16 (32 B) per lane per expert tile
        const uint4* b0p = (const uint4*)(w0 + k0 + kB);
        const uint4* b1p = (const uint4*)(w1 + k0 + kB);
        const uint4* b2p = (const uint4*)(w2 + k0 + kB);
        const uint4* b3p = (const uint4*)(w3 + k0 + kB);
        BF16x16 bf0, bf1, bf2, bf3;
        uint4 t;
        t = b0p[0]; bf0.u[0]=t.x; bf0.u[1]=t.y; bf0.u[2]=t.z; bf0.u[3]=t.w;
        t = b0p[1]; bf0.u[4]=t.x; bf0.u[5]=t.y; bf0.u[6]=t.z; bf0.u[7]=t.w;
        t = b1p[0]; bf1.u[0]=t.x; bf1.u[1]=t.y; bf1.u[2]=t.z; bf1.u[3]=t.w;
        t = b1p[1]; bf1.u[4]=t.x; bf1.u[5]=t.y; bf1.u[6]=t.z; bf1.u[7]=t.w;
        t = b2p[0]; bf2.u[0]=t.x; bf2.u[1]=t.y; bf2.u[2]=t.z; bf2.u[3]=t.w;
        t = b2p[1]; bf2.u[4]=t.x; bf2.u[5]=t.y; bf2.u[6]=t.z; bf2.u[7]=t.w;
        t = b3p[0]; bf3.u[0]=t.x; bf3.u[1]=t.y; bf3.u[2]=t.z; bf3.u[3]=t.w;
        t = b3p[1]; bf3.u[4]=t.x; bf3.u[5]=t.y; bf3.u[6]=t.z; bf3.u[7]=t.w;

        acc0 = __builtin_amdgcn_wmma_f32_16x16x32_bf16(false, af.v, false, bf0.v,
                                                       (short)0, acc0, false, false);
        acc1 = __builtin_amdgcn_wmma_f32_16x16x32_bf16(false, af.v, false, bf1.v,
                                                       (short)0, acc1, false, false);
        acc2 = __builtin_amdgcn_wmma_f32_16x16x32_bf16(false, af.v, false, bf2.v,
                                                       (short)0, acc2, false, false);
        acc3 = __builtin_amdgcn_wmma_f32_16x16x32_bf16(false, af.v, false, bf3.v,
                                                       (short)0, acc3, false, false);
    }

    // C/D layout: VGPR r, lanes 0-15 -> M=r, lanes 16-31 -> M=r+8; N = lane&15
    const int mBase = wave * 16 + 8 * hiHalf;
    #pragma unroll
    for (int r = 0; r < 8; ++r) {
        lds_logits[mBase + r][ 0 + sub] = acc0[r];
        lds_logits[mBase + r][16 + sub] = acc1[r];
        lds_logits[mBase + r][32 + sub] = acc2[r];
        lds_logits[mBase + r][48 + sub] = acc3[r];
    }
    __syncthreads();

    // ---- Phase 2: per-token softmax + top-8 (one thread per token) ----
    if (tid < TOK_PER_BLOCK) {
        float s[NEXP];
        float mx = -3.0e38f;
        #pragma unroll
        for (int e = 0; e < NEXP; ++e) { s[e] = lds_logits[tid][e]; mx = fmaxf(mx, s[e]); }
        float sum = 0.0f;
        #pragma unroll
        for (int e = 0; e < NEXP; ++e) { s[e] = __expf(s[e] - mx); sum += s[e]; }
        const float inv = __frcp_rn(sum);
        #pragma unroll
        for (int e = 0; e < NEXP; ++e) s[e] *= inv;

        unsigned long long sel = 0ull;
        float tw[TOPK]; int ti[TOPK]; float twsum = 0.0f;
        #pragma unroll
        for (int j = 0; j < TOPK; ++j) {
            float best = -1.0f; int bi = 0;
            #pragma unroll
            for (int e = 0; e < NEXP; ++e) {
                bool taken = (sel >> e) & 1ull;
                if (!taken && s[e] > best) { best = s[e]; bi = e; }
            }
            sel |= (1ull << bi);
            tw[j] = best; ti[j] = bi; twsum += best;
        }
        const float invw = __frcp_rn(twsum);
        const size_t token = (size_t)blockIdx.x * TOK_PER_BLOCK + tid;
        #pragma unroll
        for (int j = 0; j < TOPK; ++j) {
            ids_out[token * TOPK + j] = ti[j];
            w_out [token * TOPK + j] = tw[j] * invw;
        }
        // aux-loss partial: per-expert score sums within the block
        #pragma unroll
        for (int e = 0; e < NEXP; ++e) atomicAdd(&blocksum[e], s[e]);
    }
    __syncthreads();

    if (tid < NEXP) {
        const int b = (blockIdx.x * TOK_PER_BLOCK) / SEQ; // block never spans batches
        atomicAdd(&ws_acc[b * NEXP + tid], blocksum[tid]);
    }
}

// ---------------------------------------------------------------- aux -------
__global__ void moegate_aux(const float* __restrict__ ws_acc,
                            float* __restrict__ aux_out) {
    __shared__ float red[NEXP];
    int e = threadIdx.x; // 64 threads
    float v = 0.0f;
    #pragma unroll
    for (int b = 0; b < NBATCH; ++b) {
        float m = ws_acc[b * NEXP + e] * (1.0f / (float)SEQ);
        v += m * m;
    }
    red[e] = v;
    __syncthreads();
    if (e == 0) {
        float t = 0.0f;
        #pragma unroll
        for (int i = 0; i < NEXP; ++i) t += red[i];
        *aux_out = (t / (float)NBATCH) * (float)NEXP * 0.01f;
    }
}

// ---------------------------------------------------------------- launch ----
extern "C" void kernel_launch(void* const* d_in, const int* in_sizes, int n_in,
                              void* d_out, int out_size, void* d_ws, size_t ws_size,
                              hipStream_t stream) {
    const float* h = (const float*)d_in[0];   // [4,4096,2048] f32
    const float* W = (const float*)d_in[1];   // [64,2048] f32

    // d_out: ids [16384*8] (int bits), weights [16384*8] f32, aux [1] f32
    int*   ids_out = (int*)d_out;
    float* w_out   = (float*)d_out + (size_t)TOKENS * TOPK;
    float* aux_out = w_out + (size_t)TOKENS * TOPK;

    // ws: [0, 256KB) W as bf16; then 4*64 f32 accumulators
    unsigned short* wbf   = (unsigned short*)d_ws;
    float*          wsacc = (float*)((char*)d_ws + (size_t)NEXP * DIM * sizeof(unsigned short));

    moegate_prep<<<(NEXP * DIM + 255) / 256, 256, 0, stream>>>(W, wbf, wsacc);
    moegate_main<<<TOKENS / TOK_PER_BLOCK, 256, 0, stream>>>(h, wbf, ids_out, w_out, wsacc);
    moegate_aux<<<1, NEXP, 0, stream>>>(wsacc, aux_out);
}
